// UnimpNet_38723425140800
// MI455X (gfx1250) — matrix operand order repaired and checked by reading, not compiled, not run on hardware
//
#include <hip/hip_runtime.h>
#include <hip/hip_bf16.h>
#include <stdint.h>

#define NN 100000
#define EE 1600000
#define FF 256
#define HH 2
#define NCLS 40
#define DD 64

typedef __attribute__((ext_vector_type(16))) __bf16 v16bf;
typedef __attribute__((ext_vector_type(8)))  float  v8f;

// ---------------------------------------------------------------------------
// helpers
// ---------------------------------------------------------------------------
__device__ __forceinline__ void atomicMaxF(float* addr, float v) {
    if (v >= 0.0f) atomicMax((int*)addr, __float_as_int(v));
    else           atomicMin((unsigned int*)addr, __float_as_uint(v));
}
__device__ __forceinline__ float sigmoidf_(float x) { return 1.0f / (1.0f + __expf(-x)); }

// ---------------------------------------------------------------------------
// fill
// ---------------------------------------------------------------------------
__global__ void fillf(float* p, float val, long n) {
    long i = (long)blockIdx.x * blockDim.x + threadIdx.x;
    if (i < n) p[i] = val;
}

// ---------------------------------------------------------------------------
// MaskLabel: h0 = x + (mask ? emb[y] : 0)   (vectorized float4, F=256 -> 64 f4)
// ---------------------------------------------------------------------------
__global__ void masklabel(const float* __restrict__ x, const int* __restrict__ y,
                          const unsigned char* __restrict__ mask,
                          const float* __restrict__ emb, float* __restrict__ h0) {
    long t = (long)blockIdx.x * blockDim.x + threadIdx.x;
    if (t >= (long)NN * 64) return;
    int node = (int)(t >> 6);
    int c4   = (int)(t & 63);
    float4 xv = ((const float4*)x)[t];
    if (mask[node]) {
        float4 e = ((const float4*)(emb + (long)y[node] * FF))[c4];
        xv.x += e.x; xv.y += e.y; xv.z += e.z; xv.w += e.w;
    }
    ((float4*)h0)[t] = xv;
}

// ---------------------------------------------------------------------------
// Pack fp32 weight W[K, Morig] row-major into WMMA B-fragment order (bf16):
// tile (kt,nt): lane L holds col = nt*16+(L&15), K = kt*32 + 16*(L>=16) + j.
// Stored contiguously: Bp[((kt*NT+nt)*32 + lane)*16 + j].  K % 32 == 0.
// ---------------------------------------------------------------------------
__global__ void pack_w(const float* __restrict__ W, __bf16* __restrict__ Bp,
                       int K, int Morig, int Mp) {
    int NT = Mp >> 4;
    int total = (K >> 5) * NT * 32;
    int t = blockIdx.x * blockDim.x + threadIdx.x;
    if (t >= total) return;
    int lane = t & 31;
    int tile = t >> 5;
    int nt = tile % NT;
    int kt = tile / NT;
    int col   = nt * 16 + (lane & 15);
    int kbase = kt * 32 + ((lane >> 4) << 4);
    __bf16* dst = Bp + ((long)tile * 32 + lane) * 16;
#pragma unroll
    for (int j = 0; j < 16; ++j) {
        float v = (col < Morig) ? W[(long)(kbase + j) * Morig + col] : 0.0f;
        dst[j] = (__bf16)v;
    }
}

// ---------------------------------------------------------------------------
// WMMA GEMM: C[nrows, Mp] = A[nrows, K](f32->bf16) @ Bp(packed bf16) + bias
// One wave per 16-row strip; NTW column tiles per wave. wave32, EXEC all-1s.
// All NTW B fragments are fetched into distinct registers before the WMMA
// chain so loads issue as one clause and overlap the XDL pipe.
// ---------------------------------------------------------------------------
template <int NTW>
__global__ void gemm_wmma(const float* __restrict__ A, const __bf16* __restrict__ Bp,
                          const float* __restrict__ bias, float* __restrict__ C,
                          int nrows, int K, int Mp, int Morig) {
    const int lane = threadIdx.x & 31;
    const int wave = threadIdx.x >> 5;
    const int rowTile = blockIdx.x * (blockDim.x >> 5) + wave;
    const int rowBase = rowTile * 16;
    if (rowBase + 16 > nrows) return;           // uniform per wave -> EXEC stays full
    const int NT = Mp >> 4;
    const int colTile0 = blockIdx.y * NTW;
    const int m  = lane & 15;
    const int hi = lane >> 4;

    v8f acc[NTW];
#pragma unroll
    for (int t = 0; t < NTW; ++t)
#pragma unroll
        for (int r = 0; r < 8; ++r) acc[t][r] = 0.0f;

    const float* arow = A + (long)(rowBase + m) * K;
    for (int k0 = 0; k0 < K; k0 += 32) {
        const int kt = k0 >> 5;
        // issue all B-fragment loads first (distinct regs -> one clause)
        v16bf bfr[NTW];
#pragma unroll
        for (int t = 0; t < NTW; ++t)
            bfr[t] = *(const v16bf*)(Bp + ((long)(kt * NT + colTile0 + t) * 32 + lane) * 16);

        // A fragment: one contiguous 16-float run per lane (k = 16*hi + j), f32 -> bf16
        const float* ap = arow + k0 + hi * 16;
        float4 a0 = *(const float4*)(ap);
        float4 a1 = *(const float4*)(ap + 4);
        float4 a2 = *(const float4*)(ap + 8);
        float4 a3 = *(const float4*)(ap + 12);
        v16bf af;
        af[0]=(__bf16)a0.x; af[1]=(__bf16)a0.y; af[2]=(__bf16)a0.z; af[3]=(__bf16)a0.w;
        af[4]=(__bf16)a1.x; af[5]=(__bf16)a1.y; af[6]=(__bf16)a1.z; af[7]=(__bf16)a1.w;
        af[8]=(__bf16)a2.x; af[9]=(__bf16)a2.y; af[10]=(__bf16)a2.z; af[11]=(__bf16)a2.w;
        af[12]=(__bf16)a3.x; af[13]=(__bf16)a3.y; af[14]=(__bf16)a3.z; af[15]=(__bf16)a3.w;

#pragma unroll
        for (int t = 0; t < NTW; ++t)
            acc[t] = __builtin_amdgcn_wmma_f32_16x16x32_bf16(
                false, af, false, bfr[t], (short)0, acc[t], false, false);
    }
#pragma unroll
    for (int t = 0; t < NTW; ++t) {
        const int col = (colTile0 + t) * 16 + m;
        const float bv = (col < Morig) ? bias[col] : 0.0f;
#pragma unroll
        for (int r = 0; r < 8; ++r) {
            const int row = rowBase + r + hi * 8;
            C[(long)row * Mp + col] = acc[t][r] + bv;
        }
    }
}

// ---------------------------------------------------------------------------
// Edge: alpha[e,h] = dot(q[dst,h,:], k[src,h,:]) / sqrt(C); atomic segment-max.
// ---------------------------------------------------------------------------
template <int C>
__global__ void edge_alpha(const float* __restrict__ q, const float* __restrict__ k,
                           const int* __restrict__ src, const int* __restrict__ dst,
                           float* __restrict__ alpha, float* __restrict__ amax, int ld) {
    int e = blockIdx.x * blockDim.x + threadIdx.x;
    if (e >= EE) return;
    int s = src[e], d = dst[e];
    const float* qd = q + (long)d * ld;
    const float* ks = k + (long)s * ld;
    const float scale = rsqrtf((float)C);
#pragma unroll
    for (int h = 0; h < HH; ++h) {
        float acc = 0.0f;
#pragma unroll
        for (int i = 0; i < C; i += 4) {
            float4 qv = *(const float4*)(qd + h * C + i);
            float4 kv = *(const float4*)(ks + h * C + i);
            acc += qv.x * kv.x + qv.y * kv.y + qv.z * kv.z + qv.w * kv.w;
        }
        acc *= scale;
        alpha[(long)e * HH + h] = acc;
        atomicMaxF(&amax[(long)d * HH + h], acc);
    }
}

// ---------------------------------------------------------------------------
// Edge: ex = exp(alpha - amax[dst]) (in place); den[dst] += ex.
// ---------------------------------------------------------------------------
__global__ void edge_exp(float* __restrict__ alpha, const float* __restrict__ amax,
                         float* __restrict__ den, const int* __restrict__ dst) {
    int e = blockIdx.x * blockDim.x + threadIdx.x;
    if (e >= EE) return;
    int d = dst[e];
#pragma unroll
    for (int h = 0; h < HH; ++h) {
        float ex = __expf(alpha[(long)e * HH + h] - amax[(long)d * HH + h]);
        alpha[(long)e * HH + h] = ex;
        atomicAdd(&den[(long)d * HH + h], ex);
    }
}

// ---------------------------------------------------------------------------
// Edge scatter: out[dst, i] += (ex/den[dst,h]) * v[src, i]; one thread/(e,i).
// ---------------------------------------------------------------------------
template <int C>
__global__ void edge_scatter(const float* __restrict__ ex, const float* __restrict__ den,
                             const float* __restrict__ v, const int* __restrict__ src,
                             const int* __restrict__ dst, float* __restrict__ out, int ld) {
    long t = (long)blockIdx.x * blockDim.x + threadIdx.x;
    if (t >= (long)EE * (HH * C)) return;
    int e = (int)(t / (HH * C));
    int i = (int)(t % (HH * C));
    int h = i / C;
    int d = dst[e], s = src[e];
    float w = ex[(long)e * HH + h] / den[(long)d * HH + h];
    atomicAdd(&out[(long)d * ld + i], w * v[(long)s * ld + i]);
}

// ---------------------------------------------------------------------------
// Gated skip (beta) combine, so=64 concat case. One wave32 per node.
// ---------------------------------------------------------------------------
__global__ void combine64(const float* __restrict__ out, const float* __restrict__ xr,
                          const float* __restrict__ wbeta, float* __restrict__ hdst) {
    int lane = threadIdx.x & 31;
    int node = blockIdx.x * (blockDim.x >> 5) + (threadIdx.x >> 5);
    if (node >= NN) return;
    const float* o = out + (long)node * 64;
    const float* x = xr  + (long)node * 64;
    float ov[2], xv[2];
    float part = 0.0f;
#pragma unroll
    for (int t = 0; t < 2; ++t) {
        int j = lane + 32 * t;
        ov[t] = o[j]; xv[t] = x[j];
        part += ov[t] * wbeta[j] + xv[t] * wbeta[64 + j] + (ov[t] - xv[t]) * wbeta[128 + j];
    }
    for (int off = 16; off; off >>= 1) part += __shfl_xor(part, off, 32);
    float beta = sigmoidf_(part);
#pragma unroll
    for (int t = 0; t < 2; ++t) {
        int j = lane + 32 * t;
        hdst[(long)node * 64 + j] = beta * xv[t] + (1.0f - beta) * ov[t];
    }
}

// ---------------------------------------------------------------------------
// Layer-3 combine: mean over heads (C=40, out ld=80), xr ld=48 (padded),
// so=40; writes final logits [N,40].
// ---------------------------------------------------------------------------
__global__ void combine40(const float* __restrict__ out, const float* __restrict__ xr,
                          const float* __restrict__ wbeta, float* __restrict__ hdst) {
    int lane = threadIdx.x & 31;
    int node = blockIdx.x * (blockDim.x >> 5) + (threadIdx.x >> 5);
    if (node >= NN) return;
    const float* o = out + (long)node * 80;
    const float* x = xr  + (long)node * 48;
    float om[2] = {0.f, 0.f}, xv[2] = {0.f, 0.f};
    float part = 0.0f;
#pragma unroll
    for (int t = 0; t < 2; ++t) {
        int j = lane + 32 * t;
        if (j < 40) {
            om[t] = 0.5f * (o[j] + o[40 + j]);
            xv[t] = x[j];
            part += om[t] * wbeta[j] + xv[t] * wbeta[40 + j] + (om[t] - xv[t]) * wbeta[80 + j];
        }
    }
    for (int off = 16; off; off >>= 1) part += __shfl_xor(part, off, 32);
    float beta = sigmoidf_(part);
#pragma unroll
    for (int t = 0; t < 2; ++t) {
        int j = lane + 32 * t;
        if (j < 40) hdst[(long)node * 40 + j] = beta * xv[t] + (1.0f - beta) * om[t];
    }
}

// ---------------------------------------------------------------------------
// LayerNorm(64) + ReLU. One wave32 per node, 2 elements per lane.
// ---------------------------------------------------------------------------
__global__ void ln_relu(const float* __restrict__ in, const float* __restrict__ g,
                        const float* __restrict__ b, float* __restrict__ out) {
    int lane = threadIdx.x & 31;
    int node = blockIdx.x * (blockDim.x >> 5) + (threadIdx.x >> 5);
    if (node >= NN) return;
    float v0 = in[(long)node * 64 + lane];
    float v1 = in[(long)node * 64 + lane + 32];
    float s = v0 + v1;
    for (int off = 16; off; off >>= 1) s += __shfl_xor(s, off, 32);
    float mean = s * (1.0f / 64.0f);
    float d0 = v0 - mean, d1 = v1 - mean;
    float vs = d0 * d0 + d1 * d1;
    for (int off = 16; off; off >>= 1) vs += __shfl_xor(vs, off, 32);
    float inv = rsqrtf(vs * (1.0f / 64.0f) + 1e-5f);
    float r0 = d0 * inv * g[lane] + b[lane];
    float r1 = d1 * inv * g[lane + 32] + b[lane + 32];
    out[(long)node * 64 + lane]      = fmaxf(r0, 0.0f);
    out[(long)node * 64 + lane + 32] = fmaxf(r1, 0.0f);
}

// ---------------------------------------------------------------------------
// launch
// ---------------------------------------------------------------------------
extern "C" void kernel_launch(void* const* d_in, const int* in_sizes, int n_in,
                              void* d_out, int out_size, void* d_ws, size_t ws_size,
                              hipStream_t stream) {
    (void)in_sizes; (void)n_in; (void)out_size; (void)ws_size;
    const float* x          = (const float*)d_in[0];
    const int*   y          = (const int*)d_in[1];
    const int*   src        = (const int*)d_in[2];
    const int*   dst        = src + EE;
    const unsigned char* mk = (const unsigned char*)d_in[3];
    const float* emb        = (const float*)d_in[4];
    const float* WQ[3] = {(const float*)d_in[5],  (const float*)d_in[14], (const float*)d_in[23]};
    const float* BQ[3] = {(const float*)d_in[6],  (const float*)d_in[15], (const float*)d_in[24]};
    const float* WK[3] = {(const float*)d_in[7],  (const float*)d_in[16], (const float*)d_in[25]};
    const float* BK[3] = {(const float*)d_in[8],  (const float*)d_in[17], (const float*)d_in[26]};
    const float* WV[3] = {(const float*)d_in[9],  (const float*)d_in[18], (const float*)d_in[27]};
    const float* BV[3] = {(const float*)d_in[10], (const float*)d_in[19], (const float*)d_in[28]};
    const float* WS[3] = {(const float*)d_in[11], (const float*)d_in[20], (const float*)d_in[29]};
    const float* BS[3] = {(const float*)d_in[12], (const float*)d_in[21], (const float*)d_in[30]};
    const float* WB[3] = {(const float*)d_in[13], (const float*)d_in[22], (const float*)d_in[31]};
    const float* n_g[2] = {(const float*)d_in[32], (const float*)d_in[34]};
    const float* n_b[2] = {(const float*)d_in[33], (const float*)d_in[35]};

    // bump allocator over workspace
    char* p = (char*)d_ws;
    auto alloc = [&](size_t bytes) -> void* {
        void* r = (void*)p;
        p += (bytes + 255) & ~(size_t)255;
        return r;
    };
    float* h0   = (float*)alloc((size_t)NN * FF * 4);
    float* qb   = (float*)alloc((size_t)NN * 80 * 4);
    float* kb   = (float*)alloc((size_t)NN * 80 * 4);
    float* vb   = (float*)alloc((size_t)NN * 80 * 4);
    float* xrb  = (float*)alloc((size_t)NN * 64 * 4);
    float* attn = (float*)alloc((size_t)NN * 80 * 4);
    float* hA   = (float*)alloc((size_t)NN * 64 * 4);
    float* hB   = (float*)alloc((size_t)NN * 64 * 4);
    float* alph = (float*)alloc((size_t)EE * HH * 4);
    float* amax = (float*)alloc((size_t)NN * HH * 4);
    float* den  = (float*)alloc((size_t)NN * HH * 4);
    __bf16* Bpq = (__bf16*)alloc((size_t)256 * 96 * 2);
    __bf16* Bpk = (__bf16*)alloc((size_t)256 * 96 * 2);
    __bf16* Bpv = (__bf16*)alloc((size_t)256 * 96 * 2);
    __bf16* Bps = (__bf16*)alloc((size_t)256 * 96 * 2);

    const int TB = 256;
    const int gemm_gx = (NN / 16 + 7) / 8;     // 782 blocks of 8 waves
    const int node_gx = (NN + 7) / 8;          // wave-per-node kernels
    const int edge_gx = (EE + TB - 1) / TB;

    // MaskLabel
    masklabel<<<((long)NN * 64 + TB - 1) / TB, TB, 0, stream>>>(x, y, mk, emb, h0);

    for (int L = 0; L < 3; ++L) {
        const int K     = (L == 0) ? 256 : 64;
        const int C     = (L == 2) ? 40 : 32;
        const int ld    = HH * C;                 // 64 or 80
        const int Mskip = (L == 2) ? 48 : 64;     // padded
        const int MoS   = (L == 2) ? 40 : 64;
        const float* Ain = (L == 0) ? h0 : hB;

        // pack weights (bf16, WMMA B layout)
        int pt_qkv = (K >> 5) * (ld >> 4) * 32;
        int pt_s   = (K >> 5) * (Mskip >> 4) * 32;
        pack_w<<<(pt_qkv + TB - 1) / TB, TB, 0, stream>>>(WQ[L], Bpq, K, ld, ld);
        pack_w<<<(pt_qkv + TB - 1) / TB, TB, 0, stream>>>(WK[L], Bpk, K, ld, ld);
        pack_w<<<(pt_qkv + TB - 1) / TB, TB, 0, stream>>>(WV[L], Bpv, K, ld, ld);
        pack_w<<<(pt_s + TB - 1) / TB, TB, 0, stream>>>(WS[L], Bps, K, MoS, Mskip);

        // projections (WMMA)
        if (ld == 64) {
            gemm_wmma<4><<<dim3(gemm_gx, 1), TB, 0, stream>>>(Ain, Bpq, BQ[L], qb, NN, K, 64, 64);
            gemm_wmma<4><<<dim3(gemm_gx, 1), TB, 0, stream>>>(Ain, Bpk, BK[L], kb, NN, K, 64, 64);
            gemm_wmma<4><<<dim3(gemm_gx, 1), TB, 0, stream>>>(Ain, Bpv, BV[L], vb, NN, K, 64, 64);
        } else {
            gemm_wmma<5><<<dim3(gemm_gx, 1), TB, 0, stream>>>(Ain, Bpq, BQ[L], qb, NN, K, 80, 80);
            gemm_wmma<5><<<dim3(gemm_gx, 1), TB, 0, stream>>>(Ain, Bpk, BK[L], kb, NN, K, 80, 80);
            gemm_wmma<5><<<dim3(gemm_gx, 1), TB, 0, stream>>>(Ain, Bpv, BV[L], vb, NN, K, 80, 80);
        }
        if (Mskip == 64)
            gemm_wmma<4><<<dim3(gemm_gx, 1), TB, 0, stream>>>(Ain, Bps, BS[L], xrb, NN, K, 64, 64);
        else
            gemm_wmma<3><<<dim3(gemm_gx, 1), TB, 0, stream>>>(Ain, Bps, BS[L], xrb, NN, K, 48, 40);

        // init segment buffers
        fillf<<<((long)NN * HH + TB - 1) / TB, TB, 0, stream>>>(amax, -3.402823466e38f, (long)NN * HH);
        fillf<<<((long)NN * HH + TB - 1) / TB, TB, 0, stream>>>(den, 0.0f, (long)NN * HH);
        fillf<<<((long)NN * ld + TB - 1) / TB, TB, 0, stream>>>(attn, 0.0f, (long)NN * ld);

        // attention
        if (C == 32) {
            edge_alpha<32><<<edge_gx, TB, 0, stream>>>(qb, kb, src, dst, alph, amax, ld);
            edge_exp<<<edge_gx, TB, 0, stream>>>(alph, amax, den, dst);
            long st = (long)EE * 64;
            edge_scatter<32><<<(st + TB - 1) / TB, TB, 0, stream>>>(alph, den, vb, src, dst, attn, ld);
        } else {
            edge_alpha<40><<<edge_gx, TB, 0, stream>>>(qb, kb, src, dst, alph, amax, ld);
            edge_exp<<<edge_gx, TB, 0, stream>>>(alph, amax, den, dst);
            long st = (long)EE * 80;
            edge_scatter<40><<<(st + TB - 1) / TB, TB, 0, stream>>>(alph, den, vb, src, dst, attn, ld);
        }

        // gated combine (+ norm/relu for layers 0,1)
        if (L < 2) {
            combine64<<<node_gx, TB, 0, stream>>>(attn, xrb, WB[L], hA);
            ln_relu<<<node_gx, TB, 0, stream>>>(hA, n_g[L], n_b[L], hB);
        } else {
            combine40<<<node_gx, TB, 0, stream>>>(attn, xrb, WB[L], (float*)d_out);
        }
    }
}
